// VGAEncoder_36627481101161
// MI455X (gfx1250) — compile-verified, matrix operand order checked by reading, not run on hardware
//
#include <hip/hip_runtime.h>

// ---------------------------------------------------------------------------
// VGAE encoder: GCN(128->128) + ReLU, then fused GCN(128->[64|64]) for mu|logvar.
// Dense transforms: v_wmma_f32_16x16x32_bf16 (f32 accumulate), weights staged
// in LDS (shared by 8 waves/block). Sparse propagation: 1 wave/edge, float4
// per lane, global_atomic_add_f32.
// ---------------------------------------------------------------------------

typedef __attribute__((ext_vector_type(16))) __bf16 v16bf;
typedef __attribute__((ext_vector_type(8)))  float  v8f;

__device__ __forceinline__ unsigned short f32_to_bf16(float f) {
    unsigned int u = __float_as_uint(f);
    unsigned int r = u + 0x7FFFu + ((u >> 16) & 1u);   // round to nearest even
    return (unsigned short)(r >> 16);
}

// ---------------- degree / normalization ----------------

__global__ __launch_bounds__(256) void k_deg_init(float* __restrict__ deg, int n) {
    int i = blockIdx.x * blockDim.x + threadIdx.x;
    if (i < n) deg[i] = 1.0f;                           // self-loop
}

__global__ __launch_bounds__(256) void k_deg_count(const long long* __restrict__ dst,
                                                   float* __restrict__ deg, long long nE) {
    long long e = (long long)blockIdx.x * blockDim.x + threadIdx.x;
    if (e < nE) atomicAdd(&deg[(int)dst[e]], 1.0f);
}

__global__ __launch_bounds__(256) void k_rsqrt_inplace(float* __restrict__ d, int n) {
    int i = blockIdx.x * blockDim.x + threadIdx.x;
    if (i < n) d[i] = rsqrtf(d[i]);                     // deg >= 1 always
}

// ---------------- conversions / weight prep ----------------

__global__ __launch_bounds__(256) void k_cvt_bf16(const float* __restrict__ x,
                                                  unsigned short* __restrict__ o, long long total) {
    long long i = (long long)blockIdx.x * blockDim.x + threadIdx.x;
    if (i < total) o[i] = f32_to_bf16(x[i]);
}

// W1T[o][k] = W1[k][o];  WZT[o][k] = (o<64 ? Wmu[k][o] : Wlv[k][o-64])
__global__ __launch_bounds__(256) void k_prep_w(const float* __restrict__ W1,
                                                const float* __restrict__ Wmu,
                                                const float* __restrict__ Wlv,
                                                unsigned short* __restrict__ W1T,
                                                unsigned short* __restrict__ WZT) {
    int i = blockIdx.x * blockDim.x + threadIdx.x;
    if (i >= 128 * 128) return;
    int o = i >> 7, k = i & 127;
    W1T[i] = f32_to_bf16(W1[k * 128 + o]);
    float wz = (o < 64) ? Wmu[k * 64 + o] : Wlv[k * 64 + (o - 64)];
    WZT[i] = f32_to_bf16(wz);
}

// ---------------- WMMA GEMM: C[rows x 128] = A[rows x 128] @ B[128 x 128] ----
// A: bf16 row-major. BT: bf16, B transposed (row-major [128 out][128 k]).
// BT is staged into LDS once per block (8 waves share it). One wave computes a
// 16-row strip; A fragments held in regs, reused over 8 column tiles.

#define SB_STRIDE 136   // 128 + 8 elem pad: 272B row stride -> spread LDS banks

__global__ __launch_bounds__(256) void k_gemm_bf16_n128(const unsigned short* __restrict__ A,
                                                        const unsigned short* __restrict__ BT,
                                                        float* __restrict__ C, int rows) {
    __shared__ unsigned short sB[128 * SB_STRIDE];

    // ---- cooperative stage of BT (128x128 bf16) into LDS: 2 threads/row ----
    {
        const int t   = threadIdx.x;
        const int row = t >> 1;                // 0..127
        const int col = (t & 1) * 64;          // 0 or 64 (elements)
        const uint4* g = (const uint4*)(BT + row * 128 + col);
        uint4*       s = (uint4*)(sB + row * SB_STRIDE + col);
#pragma unroll
        for (int q = 0; q < 8; ++q) s[q] = g[q];   // 8 x 16B = 128B per thread
    }
    __syncthreads();

    const int lane = threadIdx.x & 31;
    const int wave = threadIdx.x >> 5;
    const int rt   = blockIdx.x * 8 + wave;             // 16-row tile index
    const int row0 = rt * 16;
    if (row0 < rows) {
        const int m    = lane & 15;
        const int half = lane >> 4;

        // A fragments for full K=128 (4 chunks of 32). Per ISA 16-bit A layout:
        // VGPR0..3 hold K = half*8 .. +7, VGPR4..7 hold K = 16+half*8 .. +7.
        v16bf a[4];
        const unsigned short* arow = A + (long long)(row0 + m) * 128 + half * 8;
#pragma unroll
        for (int kc = 0; kc < 4; ++kc) {
            const unsigned short* p = arow + kc * 32;
            *((uint4*)&a[kc])     = *(const uint4*)(p);
            *((uint4*)&a[kc] + 1) = *(const uint4*)(p + 16);
        }

#pragma unroll
        for (int ct = 0; ct < 8; ++ct) {
            const unsigned short* brow = sB + (ct * 16 + m) * SB_STRIDE + half * 8;
            v8f c = {};
#pragma unroll
            for (int kc = 0; kc < 4; ++kc) {
                v16bf b;
                const unsigned short* p = brow + kc * 32;
                *((uint4*)&b)     = *(const uint4*)(p);
                *((uint4*)&b + 1) = *(const uint4*)(p + 16);
                c = __builtin_amdgcn_wmma_f32_16x16x32_bf16(false, a[kc], false, b,
                                                            (short)0, c, false, false);
            }
            // C element (M = v + 8*half, N = lane&15)
            float* crow = C + (long long)row0 * 128 + ct * 16 + m;
#pragma unroll
            for (int v = 0; v < 8; ++v)
                crow[(long long)(v + 8 * half) * 128] = c[v];
        }
    }
}

// ---------------- propagation ----------------

// AGG[i] = dinv[node]^2 * H[i]   (self-loop contribution; also zero-initializes AGG)
__global__ __launch_bounds__(256) void k_selfloop(const float* __restrict__ H,
                                                  const float* __restrict__ dinv,
                                                  float* __restrict__ AGG, long long total) {
    long long i = (long long)blockIdx.x * blockDim.x + threadIdx.x;
    if (i >= total) return;
    float di = dinv[i >> 7];
    AGG[i] = H[i] * di * di;
}

// One wave per edge: AGG[dst,:] += H[src,:] * dinv[src]*dinv[dst]  (128 floats)
__global__ __launch_bounds__(256) void k_edge_prop(const float* __restrict__ H,
                                                   const float* __restrict__ dinv,
                                                   const long long* __restrict__ srcI,
                                                   const long long* __restrict__ dstI,
                                                   float* __restrict__ AGG, long long nE) {
    long long w = ((long long)blockIdx.x * blockDim.x + threadIdx.x) >> 5;
    int lane = threadIdx.x & 31;
    if (w >= nE) return;
    int s = (int)srcI[w];
    int d = (int)dstI[w];
    float nrm = dinv[s] * dinv[d];
    const float4 v = *(const float4*)(H + (long long)s * 128 + lane * 4);
    float* out = AGG + (long long)d * 128 + lane * 4;
    atomicAdd(out + 0, v.x * nrm);
    atomicAdd(out + 1, v.y * nrm);
    atomicAdd(out + 2, v.z * nrm);
    atomicAdd(out + 3, v.w * nrm);
}

// h_bf16 = bf16(relu(AGG + b1))
__global__ __launch_bounds__(256) void k_bias_relu_bf16(const float* __restrict__ AGG,
                                                        const float* __restrict__ b,
                                                        unsigned short* __restrict__ OUT,
                                                        long long total) {
    long long i = (long long)blockIdx.x * blockDim.x + threadIdx.x;
    if (i >= total) return;
    float v = AGG[i] + b[i & 127];
    OUT[i] = f32_to_bf16(v > 0.0f ? v : 0.0f);
}

// split fused z into (mu + bmu, logvar + blv) laid out flat in d_out
__global__ __launch_bounds__(256) void k_final(const float* __restrict__ Z,
                                               const float* __restrict__ bmu,
                                               const float* __restrict__ blv,
                                               float* __restrict__ out, long long halfOut) {
    long long i = (long long)blockIdx.x * blockDim.x + threadIdx.x;
    if (i >= halfOut) return;
    long long node = i >> 6;
    int j = (int)(i & 63);
    out[i]           = Z[node * 128 + j]      + bmu[j];
    out[halfOut + i] = Z[node * 128 + 64 + j] + blv[j];
}

// ---------------------------------------------------------------------------

extern "C" void kernel_launch(void* const* d_in, const int* in_sizes, int n_in,
                              void* d_out, int out_size, void* d_ws, size_t ws_size,
                              hipStream_t stream) {
    const float*     x    = (const float*)d_in[0];
    const long long* eidx = (const long long*)d_in[1];   // int64 [2, E]
    const float*     W1   = (const float*)d_in[2];
    const float*     b1   = (const float*)d_in[3];
    const float*     Wmu  = (const float*)d_in[4];
    const float*     bmu  = (const float*)d_in[5];
    const float*     Wlv  = (const float*)d_in[6];
    const float*     blv  = (const float*)d_in[7];
    float* out = (float*)d_out;

    const int       N  = in_sizes[0] / 128;          // 100000
    const long long E  = (long long)in_sizes[1] / 2; // 1600000
    const long long NF = (long long)N * 128;         // node-feature elements

    const long long* src = eidx;
    const long long* dst = eidx + E;

    // workspace layout (256B aligned)
    char* ws = (char*)d_ws;
    auto align256 = [](size_t v) { return (v + 255) & ~(size_t)255; };
    size_t off = 0;
    float*          dinv  = (float*)(ws + off);          off = align256(off + (size_t)N * 4);
    unsigned short* xh_bf = (unsigned short*)(ws + off); off = align256(off + (size_t)NF * 2);
    unsigned short* w1t   = (unsigned short*)(ws + off); off = align256(off + 128 * 128 * 2);
    unsigned short* wzt   = (unsigned short*)(ws + off); off = align256(off + 128 * 128 * 2);
    float*          bufA  = (float*)(ws + off);          off = align256(off + (size_t)NF * 4); // h_pre / z_pre
    float*          bufB  = (float*)(ws + off);          off = align256(off + (size_t)NF * 4); // agg / z_agg
    (void)ws_size; (void)n_in; (void)out_size;

    const int T = 256;
    const int blkN    = (N + T - 1) / T;
    const int blkNF   = (int)((NF + T - 1) / T);
    const int blkE    = (int)((E + T - 1) / T);
    const int blkEw   = (int)((E * 32 + T - 1) / T);     // 1 wave per edge
    const int rowTile = (N + 15) / 16;
    const int blkGemm = (rowTile + 7) / 8;               // 8 waves per block
    const long long halfOut = (long long)N * 64;
    const int blkOut  = (int)((halfOut + T - 1) / T);

    // normalization
    k_deg_init <<<blkN, T, 0, stream>>>(dinv, N);
    k_deg_count<<<blkE, T, 0, stream>>>(dst, dinv, E);
    k_rsqrt_inplace<<<blkN, T, 0, stream>>>(dinv, N);

    // prep bf16 inputs + transposed fused weights
    k_cvt_bf16<<<blkNF, T, 0, stream>>>(x, xh_bf, NF);
    k_prep_w  <<<(128 * 128 + T - 1) / T, T, 0, stream>>>(W1, Wmu, Wlv, w1t, wzt);

    // layer 1: h = relu(Ahat @ (X W1) + b1)
    k_gemm_bf16_n128<<<blkGemm, T, 0, stream>>>(xh_bf, w1t, bufA, N);
    k_selfloop <<<blkNF, T, 0, stream>>>(bufA, dinv, bufB, NF);
    k_edge_prop<<<blkEw, T, 0, stream>>>(bufA, dinv, src, dst, bufB, E);
    k_bias_relu_bf16<<<blkNF, T, 0, stream>>>(bufB, b1, xh_bf, NF);  // reuse xh_bf as h

    // layer 2 (fused mu|logvar): z = Ahat @ (h [Wmu|Wlv])
    k_gemm_bf16_n128<<<blkGemm, T, 0, stream>>>(xh_bf, wzt, bufA, N);
    k_selfloop <<<blkNF, T, 0, stream>>>(bufA, dinv, bufB, NF);
    k_edge_prop<<<blkEw, T, 0, stream>>>(bufA, dinv, src, dst, bufB, E);

    // split + bias into d_out (mu flat, then logvar flat)
    k_final<<<blkOut, T, 0, stream>>>(bufB, bmu, blv, out, halfOut);
}
